// MHA_68736656605707
// MI455X (gfx1250) — matrix operand order repaired and checked by reading, not compiled, not run on hardware
//
#include <hip/hip_runtime.h>
#include <hip/hip_bf16.h>

// ---------------------------------------------------------------------------
// MI455X (gfx1250) causal MHA forward, bf16 WMMA everywhere.
// B=2, L=2048, D=1024, H=16, HD=64.  ~52 GFLOP vs ~50MB unique traffic =>
// compute-bound: all GEMMs go through v_wmma_f32_16x16x32_bf16 (wave32).
// Strategy: one-shot bf16 conversion of x and transposed weights so every
// WMMA fragment register is a single aligned dword load (no VALU converts
// in the K-loops); 32x32 output per wave in the projection GEMMs for reuse.
// ---------------------------------------------------------------------------

#define B_DIM  2
#define L_DIM  2048
#define D_DIM  1024
#define H_DIM  16
#define HD_DIM 64

typedef __attribute__((ext_vector_type(8)))  unsigned v8u;
typedef __attribute__((ext_vector_type(16))) __bf16   v16bf;
typedef __attribute__((ext_vector_type(2)))  __bf16   v2bf;
typedef __attribute__((ext_vector_type(8)))  float    v8f;

__device__ __forceinline__ unsigned short f2bf_u16(float f) {
    __bf16 b = (__bf16)f;
    return __builtin_bit_cast(unsigned short, b);
}
__device__ __forceinline__ unsigned pack_bf(float lo, float hi) {
    v2bf p = {(__bf16)lo, (__bf16)hi};
    return __builtin_bit_cast(unsigned, p);
}

// D(16x16,f32) = A(16x32,bf16) x B(32x16,bf16) + C
__device__ __forceinline__ v8f wmma_bf16(v8u a, v8u b, v8f c) {
    return __builtin_amdgcn_wmma_f32_16x16x32_bf16(
        false, __builtin_bit_cast(v16bf, a),
        false, __builtin_bit_cast(v16bf, b),
        (short)0, c, false, false);
}

// Fragment index helpers (ISA 7.12.2, wave32):
//  A 16x32 bf16 : lane m=l&15, vgpr v holds K = (v>>2)*16 + (l>>4)*8 + (v&3)*2
//  B 32x16 bf16 : lane n=l&15, vgpr v holds K = (v>>2)*16 + (l>>4)*8 + (v&3)*2
//  C/D 16x16 f32: lane n=l&15, vgpr v holds M = v + 8*(l>>4)
__device__ __forceinline__ int a_kbase(int v, int half) {
    return ((v >> 2) << 4) + (half << 3) + ((v & 3) << 1);
}
__device__ __forceinline__ int b_kbase(int v, int half) {
    return ((v >> 2) << 4) + (half << 3) + ((v & 3) << 1);
}

// ---------------------------------------------------------------------------
// Prologue A: x (f32, [B*L, D]) -> bf16, straight copy.  float4 -> uint2.
// ---------------------------------------------------------------------------
__global__ void mha_cvt_x_kernel(const float4* __restrict__ x4,
                                 uint2* __restrict__ out2) {
    const int i = blockIdx.x * blockDim.x + threadIdx.x;
    const float4 v = x4[i];
    uint2 o;
    o.x = pack_bf(v.x, v.y);
    o.y = pack_bf(v.z, v.w);
    out2[i] = o;
}

// ---------------------------------------------------------------------------
// Prologue B: transpose-convert the four DxD weights to bf16 W^T[n][k]
// (so B-fragment K-pairs are contiguous).  LDS-tiled, coalesced both sides.
// grid = (D/16, D/16, 4), block = 256.
// ---------------------------------------------------------------------------
__global__ void mha_cvt_wt_kernel(const float* __restrict__ WQ,
                                  const float* __restrict__ WK,
                                  const float* __restrict__ WV,
                                  const float* __restrict__ Wo,
                                  unsigned short* __restrict__ wqt,
                                  unsigned short* __restrict__ wkt,
                                  unsigned short* __restrict__ wvt,
                                  unsigned short* __restrict__ wot) {
    __shared__ float tile[16][17];
    const int z = blockIdx.z;
    const float* W = (z == 0) ? WQ : (z == 1) ? WK : (z == 2) ? WV : Wo;
    unsigned short* O = (z == 0) ? wqt : (z == 1) ? wkt : (z == 2) ? wvt : wot;
    const int tx = threadIdx.x & 15, ty = threadIdx.x >> 4;
    const int k0 = blockIdx.x * 16, n0 = blockIdx.y * 16;
    tile[ty][tx] = W[(size_t)(k0 + ty) * D_DIM + n0 + tx];
    __syncthreads();
    O[(size_t)(n0 + ty) * D_DIM + k0 + tx] = f2bf_u16(tile[tx][ty]);
}

// ---------------------------------------------------------------------------
// Kernel 1: fused QKV projection, 32x32 output per wave (4 WMMAs / K-step).
// Q,K stored [B,H,L,HD]; V stored transposed [B,H,HD,L].
// grid = (B*L/32, 3*D/32), block = 32 (one wave).
// ---------------------------------------------------------------------------
__global__ void mha_qkv_kernel(const unsigned short* __restrict__ xbf,
                               const unsigned short* __restrict__ wqt,
                               const unsigned short* __restrict__ wkt,
                               const unsigned short* __restrict__ wvt,
                               unsigned short* __restrict__ qws,
                               unsigned short* __restrict__ kws,
                               unsigned short* __restrict__ vws) {
    const int lane = threadIdx.x & 31;
    const int half = lane >> 4;
    const int nn   = lane & 15;
    const int row0 = blockIdx.x * 32;            // over B*L
    const int wsel = blockIdx.y >> 5;            // 0:Q 1:K 2:V
    const int ncol = (blockIdx.y & 31) * 32;
    const unsigned short* Wt = (wsel == 0) ? wqt : (wsel == 1) ? wkt : wvt;

    const unsigned short* ar0 = xbf + (size_t)(row0 + nn) * D_DIM;
    const unsigned short* ar1 = ar0 + (size_t)16 * D_DIM;
    const unsigned short* br0 = Wt + (size_t)(ncol + nn) * D_DIM;
    const unsigned short* br1 = br0 + (size_t)16 * D_DIM;

    v8f acc00 = {}, acc01 = {}, acc10 = {}, acc11 = {};
    for (int k0 = 0; k0 < D_DIM; k0 += 32) {
        __builtin_prefetch(ar0 + k0 + 128, 0, 0);
        __builtin_prefetch(br0 + k0 + 128, 0, 0);
        v8u a0, a1, b0, b1;
#pragma unroll
        for (int v = 0; v < 8; ++v) {
            const int ka = a_kbase(v, half), kb = b_kbase(v, half);
            a0[v] = *(const unsigned*)(ar0 + k0 + ka);
            a1[v] = *(const unsigned*)(ar1 + k0 + ka);
            b0[v] = *(const unsigned*)(br0 + k0 + kb);
            b1[v] = *(const unsigned*)(br1 + k0 + kb);
        }
        acc00 = wmma_bf16(a0, b0, acc00);
        acc01 = wmma_bf16(a0, b1, acc01);
        acc10 = wmma_bf16(a1, b0, acc10);
        acc11 = wmma_bf16(a1, b1, acc11);
    }

    const int bidx = row0 / L_DIM;               // 32 | L: tile in one batch
    const int l0   = row0 % L_DIM;
    // column groups: nt=0 -> ncol+nn ; nt=1 -> ncol+16+nn
    const int dg0 = ncol + nn,      dg1 = ncol + 16 + nn;
    const int h0  = dg0 >> 6,       h1  = dg1 >> 6;
    const int d0  = dg0 & 63,       d1  = dg1 & 63;

    if (wsel < 2) {                              // uniform branch per wave
        unsigned short* o = (wsel == 0) ? qws : kws;
        const size_t c0 = ((size_t)(bidx * H_DIM + h0) * L_DIM + l0) * HD_DIM + d0;
        const size_t c1 = ((size_t)(bidx * H_DIM + h1) * L_DIM + l0) * HD_DIM + d1;
#pragma unroll
        for (int v = 0; v < 8; ++v) {
            const int m0 = v + (half << 3), m1 = m0 + 16;
            o[c0 + (size_t)m0 * HD_DIM] = f2bf_u16(acc00[v]);
            o[c1 + (size_t)m0 * HD_DIM] = f2bf_u16(acc01[v]);
            o[c0 + (size_t)m1 * HD_DIM] = f2bf_u16(acc10[v]);
            o[c1 + (size_t)m1 * HD_DIM] = f2bf_u16(acc11[v]);
        }
    } else {                                     // V transposed [B,H,HD,L]
        const size_t c0 = ((size_t)(bidx * H_DIM + h0) * HD_DIM + d0) * L_DIM + l0;
        const size_t c1 = ((size_t)(bidx * H_DIM + h1) * HD_DIM + d1) * L_DIM + l0;
#pragma unroll
        for (int v = 0; v < 8; ++v) {
            const int m0 = v + (half << 3), m1 = m0 + 16;
            vws[c0 + m0] = f2bf_u16(acc00[v]);
            vws[c1 + m0] = f2bf_u16(acc01[v]);
            vws[c0 + m1] = f2bf_u16(acc10[v]);
            vws[c1 + m1] = f2bf_u16(acc11[v]);
        }
    }
}

// ---------------------------------------------------------------------------
// Kernel 2: causal flash attention.  One wave per (16-query tile, b*h).
// Per 32-key chunk: 4 WMMAs for S = Q K^T, online softmax (shfl_xor row
// reductions in 16-lane groups), C->A relayout of P via LDS, 4 WMMAs for PV.
// grid = (L/16, B*H), block = 32.
// ---------------------------------------------------------------------------
__global__ void mha_attn_kernel(const unsigned short* __restrict__ qws,
                                const unsigned short* __restrict__ kws,
                                const unsigned short* __restrict__ vws,
                                unsigned short* __restrict__ cws) {
    __shared__ float s_p[16 * 32];
    const int lane = threadIdx.x & 31;
    const int half = lane >> 4;
    const int nn   = lane & 15;
    const int qt   = blockIdx.x;
    const int bh   = blockIdx.y;
    const size_t qkbase = (size_t)bh * L_DIM * HD_DIM;

    v8u aQ[2];
    {
        const unsigned short* qrow =
            qws + qkbase + (size_t)(qt * 16 + nn) * HD_DIM;
#pragma unroll
        for (int f = 0; f < 2; ++f)
#pragma unroll
            for (int v = 0; v < 8; ++v)
                aQ[f][v] = *(const unsigned*)(qrow + f * 32 + a_kbase(v, half));
    }

    v8f acc[4] = {v8f{}, v8f{}, v8f{}, v8f{}};
    float rmax[8], rsum[8];
#pragma unroll
    for (int v = 0; v < 8; ++v) { rmax[v] = -1e30f; rsum[v] = 0.f; }

    const int nch = (qt * 16 + 15) / 32 + 1;
    for (int c = 0; c < nch; ++c) {
        const int key0 = c * 32;
        v8f S[2];
#pragma unroll
        for (int sub = 0; sub < 2; ++sub) {
            const unsigned short* krow =
                kws + qkbase + (size_t)(key0 + sub * 16 + nn) * HD_DIM;
            v8u bK0, bK1;
#pragma unroll
            for (int v = 0; v < 8; ++v) {
                const int kb = b_kbase(v, half);
                bK0[v] = *(const unsigned*)(krow + kb);
                bK1[v] = *(const unsigned*)(krow + 32 + kb);
            }
            v8f s = {};
            s = wmma_bf16(aQ[0], bK0, s);
            s = wmma_bf16(aQ[1], bK1, s);
            S[sub] = s;
        }
#pragma unroll
        for (int sub = 0; sub < 2; ++sub)
#pragma unroll
            for (int v = 0; v < 8; ++v) {
                const int qg = qt * 16 + v + (half << 3);
                const int kg = key0 + sub * 16 + nn;
                const float sv = S[sub][v] * 0.125f;   // 1/sqrt(64)
                S[sub][v] = (kg <= qg) ? sv : -1e30f;
            }
#pragma unroll
        for (int v = 0; v < 8; ++v) {
            float mx = fmaxf(S[0][v], S[1][v]);
#pragma unroll
            for (int off = 8; off >= 1; off >>= 1)
                mx = fmaxf(mx, __shfl_xor(mx, off, 32));
            const float nm   = fmaxf(rmax[v], mx);
            const float corr = __expf(rmax[v] - nm);
            rmax[v] = nm;
            const float p0 = __expf(S[0][v] - nm);
            const float p1 = __expf(S[1][v] - nm);
            float ps = p0 + p1;
#pragma unroll
            for (int off = 8; off >= 1; off >>= 1)
                ps += __shfl_xor(ps, off, 32);
            rsum[v] = rsum[v] * corr + ps;
#pragma unroll
            for (int t = 0; t < 4; ++t) acc[t][v] *= corr;
            const int m = v + (half << 3);
            s_p[m * 32 + nn]      = p0;
            s_p[m * 32 + 16 + nn] = p1;
        }
        __syncthreads();
        v8u aP;
#pragma unroll
        for (int v = 0; v < 8; ++v) {
            const int ka = a_kbase(v, half);
            aP[v] = pack_bf(s_p[nn * 32 + ka], s_p[nn * 32 + ka + 1]);
        }
        __syncthreads();
        const size_t vbase = (size_t)bh * HD_DIM * L_DIM;
#pragma unroll
        for (int t = 0; t < 4; ++t) {
            const unsigned short* vrow =
                vws + vbase + (size_t)(t * 16 + nn) * L_DIM + key0;
            v8u bV;
#pragma unroll
            for (int v = 0; v < 8; ++v)
                bV[v] = *(const unsigned*)(vrow + b_kbase(v, half));
            acc[t] = wmma_bf16(aP, bV, acc[t]);
        }
    }
    const int b = bh / H_DIM, h = bh % H_DIM;
#pragma unroll
    for (int t = 0; t < 4; ++t)
#pragma unroll
        for (int v = 0; v < 8; ++v) {
            const int qg = qt * 16 + v + (half << 3);
            const float val = acc[t][v] / rsum[v];
            cws[(size_t)(b * L_DIM + qg) * D_DIM + h * 64 + t * 16 + nn] =
                f2bf_u16(val);
        }
}

// ---------------------------------------------------------------------------
// Kernel 3: output projection, 32x32 per wave.  out = ctx @ Wo + bo (f32).
// grid = (B*L/32, D/32), block = 32.
// ---------------------------------------------------------------------------
__global__ void mha_proj_kernel(const unsigned short* __restrict__ cws,
                                const unsigned short* __restrict__ wot,
                                const float* __restrict__ bo,
                                float* __restrict__ out) {
    const int lane = threadIdx.x & 31;
    const int half = lane >> 4;
    const int nn   = lane & 15;
    const int row0 = blockIdx.x * 32;
    const int ncol = blockIdx.y * 32;

    const unsigned short* ar0 = cws + (size_t)(row0 + nn) * D_DIM;
    const unsigned short* ar1 = ar0 + (size_t)16 * D_DIM;
    const unsigned short* br0 = wot + (size_t)(ncol + nn) * D_DIM;
    const unsigned short* br1 = br0 + (size_t)16 * D_DIM;

    v8f acc00 = {}, acc01 = {}, acc10 = {}, acc11 = {};
    for (int k0 = 0; k0 < D_DIM; k0 += 32) {
        __builtin_prefetch(ar0 + k0 + 128, 0, 0);
        __builtin_prefetch(br0 + k0 + 128, 0, 0);
        v8u a0, a1, b0, b1;
#pragma unroll
        for (int v = 0; v < 8; ++v) {
            const int ka = a_kbase(v, half), kb = b_kbase(v, half);
            a0[v] = *(const unsigned*)(ar0 + k0 + ka);
            a1[v] = *(const unsigned*)(ar1 + k0 + ka);
            b0[v] = *(const unsigned*)(br0 + k0 + kb);
            b1[v] = *(const unsigned*)(br1 + k0 + kb);
        }
        acc00 = wmma_bf16(a0, b0, acc00);
        acc01 = wmma_bf16(a0, b1, acc01);
        acc10 = wmma_bf16(a1, b0, acc10);
        acc11 = wmma_bf16(a1, b1, acc11);
    }
    const float bias0 = bo[ncol + nn];
    const float bias1 = bo[ncol + 16 + nn];
#pragma unroll
    for (int v = 0; v < 8; ++v) {
        const int m0 = v + (half << 3), m1 = m0 + 16;
        out[(size_t)(row0 + m0) * D_DIM + ncol + nn]      = acc00[v] + bias0;
        out[(size_t)(row0 + m0) * D_DIM + ncol + 16 + nn] = acc01[v] + bias1;
        out[(size_t)(row0 + m1) * D_DIM + ncol + nn]      = acc10[v] + bias0;
        out[(size_t)(row0 + m1) * D_DIM + ncol + 16 + nn] = acc11[v] + bias1;
    }
}

// ---------------------------------------------------------------------------
extern "C" void kernel_launch(void* const* d_in, const int* in_sizes, int n_in,
                              void* d_out, int out_size, void* d_ws,
                              size_t ws_size, hipStream_t stream) {
    const float* x  = (const float*)d_in[0];
    const float* WQ = (const float*)d_in[1];
    const float* WK = (const float*)d_in[2];
    const float* WV = (const float*)d_in[3];
    const float* Wo = (const float*)d_in[4];
    const float* bo = (const float*)d_in[5];

    const size_t qk_elems = (size_t)B_DIM * H_DIM * L_DIM * HD_DIM; // 4 Mi
    const size_t w_elems  = (size_t)D_DIM * D_DIM;                  // 1 Mi
    unsigned short* qws = (unsigned short*)d_ws;
    unsigned short* kws = qws + qk_elems;
    unsigned short* vws = kws + qk_elems;
    unsigned short* cws = vws + qk_elems;       // ctx  (B*L*D bf16)
    unsigned short* xbf = cws + qk_elems;       // x    (B*L*D bf16)
    unsigned short* wqt = xbf + qk_elems;       // W^T bf16, 4x
    unsigned short* wkt = wqt + w_elems;
    unsigned short* wvt = wkt + w_elems;
    unsigned short* wot = wvt + w_elems;

    // prologue conversions (one-shot per launch)
    const int n4 = (int)(qk_elems / 4);         // x as float4
    mha_cvt_x_kernel<<<n4 / 256, 256, 0, stream>>>((const float4*)x,
                                                   (uint2*)xbf);
    mha_cvt_wt_kernel<<<dim3(D_DIM / 16, D_DIM / 16, 4), 256, 0, stream>>>(
        WQ, WK, WV, Wo, wqt, wkt, wvt, wot);

    mha_qkv_kernel<<<dim3(B_DIM * L_DIM / 32, 3 * D_DIM / 32), 32, 0, stream>>>(
        xbf, wqt, wkt, wvt, qws, kws, vws);

    mha_attn_kernel<<<dim3(L_DIM / 16, B_DIM * H_DIM), 32, 0, stream>>>(
        qws, kws, vws, cws);

    mha_proj_kernel<<<dim3(B_DIM * L_DIM / 32, D_DIM / 32), 32, 0, stream>>>(
        cws, wot, bo, (float*)d_out);
}
